// GCN_backbone_70188355551856
// MI455X (gfx1250) — matrix operand order repaired and checked by reading, not compiled, not run on hardware
//
#include <hip/hip_runtime.h>
#include <hip/hip_bf16.h>

#define DIM 256
#define ROWS_PER_WAVE 8   // row tiles swept per wave, reusing B fragments in registers

typedef __attribute__((ext_vector_type(16))) __bf16 v16bf;
typedef __attribute__((ext_vector_type(8)))  __bf16 v8bf;
typedef __attribute__((ext_vector_type(8)))  float  v8f;

// float -> bf16, round-to-nearest-even (used only in memory-bound prep kernels)
static __device__ __forceinline__ __bf16 f2bf(float f) {
    unsigned u = __builtin_bit_cast(unsigned, f);
    unsigned r = (u + 0x7FFFu + ((u >> 16) & 1u)) >> 16;
    unsigned short s = (unsigned short)r;
    return __builtin_bit_cast(__bf16, s);
}

// ---------------- degree / norm ----------------
__global__ void k_init_deg(float* __restrict__ deg, int N) {
    int n = blockIdx.x * blockDim.x + threadIdx.x;
    if (n < N) deg[n] = 1.0f;   // self-loop
}

__global__ void k_count_deg(const int* __restrict__ dst, float* __restrict__ deg, int E) {
    int e = blockIdx.x * blockDim.x + threadIdx.x;
    if (e < E) atomicAdd(&deg[dst[e]], 1.0f);
}

__global__ void k_to_dinv(float* __restrict__ deg, int N) {
    int n = blockIdx.x * blockDim.x + threadIdx.x;
    if (n < N) deg[n] = rsqrtf(deg[n]);
}

// ---------------- float4 copy (embeds[0] = x) ----------------
__global__ void k_copy4(const float* __restrict__ in, float* __restrict__ out, int n4) {
    int i = blockIdx.x * blockDim.x + threadIdx.x;
    if (i < n4) ((float4*)out)[i] = ((const float4*)in)[i];
}

// ---------------- fp32 -> bf16 matrix convert (8 elems / thread) ----------------
__global__ void k_cvt_bf16(const float* __restrict__ in, __bf16* __restrict__ out, int n8) {
    int i = blockIdx.x * blockDim.x + threadIdx.x;
    if (i >= n8) return;
    const float4* p = (const float4*)(in + (size_t)i * 8);
    float4 v0 = p[0], v1 = p[1];
    v8bf o;
    o[0] = f2bf(v0.x); o[1] = f2bf(v0.y); o[2] = f2bf(v0.z); o[3] = f2bf(v0.w);
    o[4] = f2bf(v1.x); o[5] = f2bf(v1.y); o[6] = f2bf(v1.z); o[7] = f2bf(v1.w);
    *(v8bf*)(out + (size_t)i * 8) = o;
}

// ---------------- in-place ReLU + bf16 convert (layer epilogue) ----------------
__global__ void k_relu_cvt(float* __restrict__ p, __bf16* __restrict__ ob, int n8) {
    int i = blockIdx.x * blockDim.x + threadIdx.x;
    if (i >= n8) return;
    float4* q = (float4*)(p + (size_t)i * 8);
    float4 v0 = q[0], v1 = q[1];
    v0.x = fmaxf(v0.x, 0.f); v0.y = fmaxf(v0.y, 0.f);
    v0.z = fmaxf(v0.z, 0.f); v0.w = fmaxf(v0.w, 0.f);
    v1.x = fmaxf(v1.x, 0.f); v1.y = fmaxf(v1.y, 0.f);
    v1.z = fmaxf(v1.z, 0.f); v1.w = fmaxf(v1.w, 0.f);
    q[0] = v0; q[1] = v1;
    v8bf o;
    o[0] = f2bf(v0.x); o[1] = f2bf(v0.y); o[2] = f2bf(v0.z); o[3] = f2bf(v0.w);
    o[4] = f2bf(v1.x); o[5] = f2bf(v1.y); o[6] = f2bf(v1.z); o[7] = f2bf(v1.w);
    *(v8bf*)(ob + (size_t)i * 8) = o;
}

// ---------------- pack all 4 weight matrices into per-lane B-fragment layout ----
// Wpk index: (((l*8 + ks)*16 + ct)*32 + lane)*16 + e
// holds bf16( W[l][ K = ks*32 + 8*(lane>>4) + (e&7) + 16*(e>>3) ][ col = ct*16 + (lane&15) ] )
__global__ void k_pack_W(const float* __restrict__ W, __bf16* __restrict__ Wpk, int total) {
    int idx = blockIdx.x * blockDim.x + threadIdx.x;
    if (idx >= total) return;
    int e    = idx & 15;
    int lane = (idx >> 4) & 31;
    int ct   = (idx >> 9) & 15;
    int ks   = (idx >> 13) & 7;
    int l    = idx >> 16;
    int K    = ks * 32 + 8 * (lane >> 4) + (e & 7) + 16 * (e >> 3);
    int col  = ct * 16 + (lane & 15);
    Wpk[idx] = f2bf(W[(size_t)l * DIM * DIM + (size_t)K * DIM + col]);
}

// ---------------- GEMM: tmp[N,256] = A_bf16[N,256] @ W_l  (bf16 WMMA, f32 acc) ----
// One wave owns one 16-col strip and sweeps ROWS_PER_WAVE row tiles.
// All 8 B fragments (full K=256 strip, 64 VGPRs/lane) are hoisted into registers
// once; each row tile then costs only 16x b128 A-loads + 8 WMMAs (~1KB/WMMA).
// VGPR fragment layouts per CDNA5 ISA 7.12.2:
//   A (16x32 bf16): lanes 0-15 row M=lane hold K={0..7,16..23}; lanes 16-31 K={8..15,24..31}
//   C/D (16x16 f32): VGPR r: lanes 0-15 -> M=r, lanes 16-31 -> M=r+8
__global__ void __launch_bounds__(128)
k_gemm_bf16(const __bf16* __restrict__ A, const __bf16* __restrict__ Bpk,
            float* __restrict__ outp, int N) {
    const int wave = threadIdx.x >> 5;
    const int lane = threadIdx.x & 31;
    const int wid  = blockIdx.x * 4 + wave;
    const int rowTiles  = (N + 15) >> 4;
    const int rowGroups = (rowTiles + ROWS_PER_WAVE - 1) / ROWS_PER_WAVE;
    const int rg = wid >> 4;            // row-tile group
    const int ct = wid & 15;            // 16 col tiles (256/16)
    if (rg >= rowGroups) return;        // uniform across the wave

    const int half = lane >> 4;
    const int l16  = lane & 15;

    // hoist all 8 B fragments for this column strip into registers
    const __bf16* bfrag = Bpk + ((size_t)ct * 32 + lane) * 16;
    v16bf bfr[8];
    #pragma unroll
    for (int ks = 0; ks < 8; ++ks) {
        const __bf16* bp = bfrag + (size_t)ks * (16 * 32 * 16);
        v8bf b0 = *(const v8bf*)(bp);
        v8bf b1 = *(const v8bf*)(bp + 8);
        bfr[ks] = __builtin_shufflevector(b0, b1, 0,1,2,3,4,5,6,7,8,9,10,11,12,13,14,15);
    }

    const int rtBase = rg * ROWS_PER_WAVE;
    for (int rs = 0; rs < ROWS_PER_WAVE; ++rs) {
        const int rt = rtBase + rs;
        if (rt >= rowTiles) break;      // uniform across the wave
        int mrow = rt * 16 + l16;
        if (mrow >= N) mrow = N - 1;    // clamp loads; keeps EXEC all-1s for WMMA
        const __bf16* arow = A + (size_t)mrow * DIM + half * 8;

        v8f acc = {};
        #pragma unroll
        for (int ks = 0; ks < 8; ++ks) {
            v8bf a0 = *(const v8bf*)(arow + ks * 32);       // K = ks*32 + 8*half + 0..7
            v8bf a1 = *(const v8bf*)(arow + ks * 32 + 16);  // K = ks*32 + 16 + 8*half + 0..7
            v16bf a = __builtin_shufflevector(a0, a1, 0,1,2,3,4,5,6,7,8,9,10,11,12,13,14,15);
            acc = __builtin_amdgcn_wmma_f32_16x16x32_bf16(
                      false, a, false, bfr[ks], (short)0, acc, false, false);
        }
        #pragma unroll
        for (int r = 0; r < 8; ++r) {
            int m = rt * 16 + r + half * 8;
            if (m < N) outp[(size_t)m * DIM + ct * 16 + l16] = acc[r];
        }
    }
}

// ---------------- aggregation ----------------
// out[n,:] = bias + tmp[n,:] * dinv[n]^2   (self-loop folded in; rewrites whole slice)
__global__ void k_init_agg(const float* __restrict__ tmp, const float* __restrict__ dinv,
                           const float* __restrict__ bias, float* __restrict__ outp, int N) {
    int idx = blockIdx.x * blockDim.x + threadIdx.x;   // over N*64 float4s
    if (idx >= N * 64) return;
    int n  = idx >> 6;
    int d4 = (idx & 63) << 2;
    float w = dinv[n]; w *= w;
    float4 v  = *(const float4*)(tmp  + (size_t)n * DIM + d4);
    float4 bb = *(const float4*)(bias + d4);
    float4 r;
    r.x = fmaf(v.x, w, bb.x); r.y = fmaf(v.y, w, bb.y);
    r.z = fmaf(v.z, w, bb.z); r.w = fmaf(v.w, w, bb.w);
    *(float4*)(outp + (size_t)n * DIM + d4) = r;
}

// out[dst,:] += tmp[src,:] * dinv[src]*dinv[dst]   (4 edges per 256-thread block)
__global__ void __launch_bounds__(256)
k_scatter(const float* __restrict__ tmp, const int* __restrict__ src,
          const int* __restrict__ dst, const float* __restrict__ dinv,
          float* __restrict__ outp, int E) {
    int t  = threadIdx.x;
    int e  = blockIdx.x * 4 + (t >> 6);
    if (e >= E) return;
    int d4 = (t & 63) << 2;
    int s = src[e], d = dst[e];
    float w = dinv[s] * dinv[d];
    float4 v = *(const float4*)(tmp + (size_t)s * DIM + d4);
    float* o = outp + (size_t)d * DIM + d4;
    atomicAdd(o + 0, v.x * w);
    atomicAdd(o + 1, v.y * w);
    atomicAdd(o + 2, v.z * w);
    atomicAdd(o + 3, v.w * w);
}

extern "C" void kernel_launch(void* const* d_in, const int* in_sizes, int n_in,
                              void* d_out, int out_size, void* d_ws, size_t ws_size,
                              hipStream_t stream) {
    const float* x  = (const float*)d_in[0];
    const int*   ei = (const int*)d_in[1];    // [2, E] int32 (JAX x64 disabled)
    const float* W  = (const float*)d_in[2];  // [4, 256, 256]
    const float* b  = (const float*)d_in[3];  // [4, 256]

    const int N = in_sizes[0] / DIM;
    const int E = in_sizes[1] / 2;
    const int* srcA = ei;
    const int* dstA = ei + E;

    float* out = (float*)d_out;

    // workspace: dinv [N] | tmp fp32 [N*256] | Hbf bf16 [N*256] | Wpk bf16 [4*256*256]
    char*  ws   = (char*)d_ws;
    size_t off  = 0;
    float* dinv = (float*)(ws + off);
    off += (((size_t)N * 4) + 255) & ~(size_t)255;
    float* tmp  = (float*)(ws + off);
    off += (size_t)N * DIM * 4;
    __bf16* Hbf = (__bf16*)(ws + off);
    off += (size_t)N * DIM * 2;
    __bf16* Wpk = (__bf16*)(ws + off);

    const int n4 = N * (DIM / 4);
    const int n8 = N * (DIM / 8);
    const size_t slice = (size_t)N * DIM;

    // normalization: deg -> dinv
    k_init_deg <<<(N + 255) / 256, 256, 0, stream>>>(dinv, N);
    k_count_deg<<<(E + 255) / 256, 256, 0, stream>>>(dstA, dinv, E);
    k_to_dinv  <<<(N + 255) / 256, 256, 0, stream>>>(dinv, N);

    // embeds[0] = x ; Hbf = bf16(x) ; pack weights once
    k_copy4   <<<(n4 + 255) / 256, 256, 0, stream>>>(x, out, n4);
    k_cvt_bf16<<<(n8 + 255) / 256, 256, 0, stream>>>(x, Hbf, n8);
    const int wTotal = 4 * DIM * DIM;
    k_pack_W  <<<(wTotal + 255) / 256, 256, 0, stream>>>(W, Wpk, wTotal);

    const int rowTiles  = (N + 15) / 16;
    const int rowGroups = (rowTiles + ROWS_PER_WAVE - 1) / ROWS_PER_WAVE;
    const int gemmBlks  = (rowGroups * 16 + 3) / 4;

    for (int l = 0; l < 4; ++l) {
        float* oSlice = out + (size_t)(l + 1) * slice;
        k_gemm_bf16<<<gemmBlks, 128, 0, stream>>>(Hbf, Wpk + (size_t)l * DIM * DIM, tmp, N);
        k_init_agg <<<(N * 64 + 255) / 256, 256, 0, stream>>>(tmp, dinv, b + (size_t)l * DIM, oSlice, N);
        k_scatter  <<<(E + 3) / 4, 256, 0, stream>>>(tmp, srcA, dstA, dinv, oSlice, E);
        if (l < 3) k_relu_cvt<<<(n8 + 255) / 256, 256, 0, stream>>>(oSlice, Hbf, n8);
    }
}
